// App_Classifier_19928648253913
// MI455X (gfx1250) — matrix operand order repaired and checked by reading, not compiled, not run on hardware
//
#include <hip/hip_runtime.h>

typedef __attribute__((ext_vector_type(16))) __bf16 v16bf;
typedef __attribute__((ext_vector_type(8)))  float  v8f;

// ---------------------------------------------------------------------------
// Utility kernels
// ---------------------------------------------------------------------------
__global__ void zero_kernel(float* __restrict__ p, long long n) {
    long long i = (long long)blockIdx.x * blockDim.x + threadIdx.x;
    long long stride = (long long)gridDim.x * blockDim.x;
    for (; i < n; i += stride) p[i] = 0.0f;
}

__global__ void degree_kernel(const int* __restrict__ src, const int* __restrict__ dst,
                              float* __restrict__ odeg, float* __restrict__ ideg, int E) {
    int i = blockIdx.x * blockDim.x + threadIdx.x;
    if (i < E) {
        atomicAdd(&odeg[src[i]], 1.0f);
        atomicAdd(&ideg[dst[i]], 1.0f);
    }
}

__global__ void rsqrt_kernel(float* __restrict__ p, long long n) {
    long long i = (long long)blockIdx.x * blockDim.x + threadIdx.x;
    if (i < n) p[i] = rsqrtf(fmaxf(p[i], 1.0f));
}

__global__ void count_kernel(const int* __restrict__ gid, float* __restrict__ cnt, int N) {
    int i = blockIdx.x * blockDim.x + threadIdx.x;
    if (i < N) atomicAdd(&cnt[gid[i]], 1.0f);
}

// One block per edge: agg[dst] += x[src] * out_is[src] * in_is[dst]
// (in-degree normalization folded in by linearity of the segment sum)
__global__ void scatter_kernel(const float* __restrict__ x, const int* __restrict__ src,
                               const int* __restrict__ dst, const float* __restrict__ oscale,
                               const float* __restrict__ iscale, float* __restrict__ agg,
                               int dim, int xstride, int astride) {
    int e = blockIdx.x;
    int s = src[e];
    int d = dst[e];
    float sc = oscale[s] * iscale[d];
    const float* xs = x + (size_t)s * xstride;
    float* ad = agg + (size_t)d * astride;
    for (int f = threadIdx.x; f < dim; f += blockDim.x)
        atomicAdd(&ad[f], xs[f] * sc);
}

// One block per node: feat[gid[n]*fstride + foff + f] += x[n*xstride + f]
__global__ void pool_kernel(const float* __restrict__ x, const int* __restrict__ gid,
                            float* __restrict__ feat, int dim, int xstride,
                            int fstride, int foff) {
    int n = blockIdx.x;
    int g = gid[n];
    const float* xn = x + (size_t)n * xstride;
    float* fg = feat + (size_t)g * fstride + foff;
    for (int f = threadIdx.x; f < dim; f += blockDim.x)
        atomicAdd(&fg[f], xn[f]);
}

__global__ void divide_kernel(float* __restrict__ feat, const float* __restrict__ cnt,
                              long long total, int fdim, int fstride) {
    long long i = (long long)blockIdx.x * blockDim.x + threadIdx.x;
    if (i < total) {
        int g = (int)(i / fdim);
        int j = (int)(i % fdim);
        feat[(size_t)g * fstride + j] /= fmaxf(cnt[g], 1.0f);
    }
}

// ---------------------------------------------------------------------------
// Weight swizzle: fp32 W[K,Nc] -> bf16 B fragments in contiguous k order (same as A in GEMM).
// Element e of lane L (half hf=L>>4, col lr=L&15) of k-tile kt / n-tile tn is
// W[kt*32 + 16*hf + e, tn*16 + lr], zero-padded outside [K,Nc].
// Each lane's 16 bf16 values are contiguous (32B) -> two b128 loads in GEMM.
// ---------------------------------------------------------------------------
__global__ void swizzle_w_kernel(const float* __restrict__ W, __bf16* __restrict__ Wsw,
                                 int K, int Nc, int ntp, long long total) {
    long long idx = (long long)blockIdx.x * blockDim.x + threadIdx.x;
    if (idx >= total) return;
    int e    = (int)(idx & 15);
    int lane = (int)((idx >> 4) & 31);
    long long t = idx >> 9;
    int tn = (int)(t % ntp);
    int kt = (int)(t / ntp);
    int k = kt * 32 + 16 * (lane >> 4) + e;
    int n = tn * 16 + (lane & 15);
    float v = (k < K && n < Nc) ? W[(size_t)k * Nc + n] : 0.0f;
    Wsw[idx] = (__bf16)v;
}

// ---------------------------------------------------------------------------
// WMMA GEMM, one wave per 16-row block. A fp32 [M x NK*32] (zero/finite pad),
// W pre-swizzled bf16, C fp32 [M x ldc] (only cols < Nc written), bias+ReLU
// fused. A row-block converted to bf16 fragments in registers ONCE, then the
// wave sweeps all N-tiles: steady state = 2x global_load_b128 + 1x wmma.
// ---------------------------------------------------------------------------
template <int NK>
__global__ void gemm_rows_wmma(const float* __restrict__ A, const __bf16* __restrict__ Wsw,
                               const float* __restrict__ bias, float* __restrict__ C,
                               int M, int Nc, int ntp, int ldc, int relu)
{
    const int lda = NK * 32;
    int wavesPerBlock = blockDim.x >> 5;
    int wave = blockIdx.x * wavesPerBlock + (threadIdx.x >> 5);
    int mt = (M + 15) >> 4;
    if (wave >= mt) return;                 // wave-uniform exit, EXEC stays full
    int lane = threadIdx.x & 31;
    int hf = lane >> 4;
    int lr = lane & 15;
    int m0 = wave << 4;
    int am = m0 + lr;
    const float* Arow = A + (size_t)(am < M ? am : 0) * lda;

    // ---- load + convert full 16 x (NK*32) A block into registers
    v16bf af[NK];
    #pragma unroll
    for (int kt = 0; kt < NK; ++kt) {
        const float* p0 = Arow + kt * 32 + 16 * hf;
        float4 f0 = *(const float4*)(p0);
        float4 f1 = *(const float4*)(p0 + 4);
        float4 f2 = *(const float4*)(p0 + 8);
        float4 f3 = *(const float4*)(p0 + 12);
        af[kt][0]  = (__bf16)f0.x; af[kt][1]  = (__bf16)f0.y;
        af[kt][2]  = (__bf16)f0.z; af[kt][3]  = (__bf16)f0.w;
        af[kt][4]  = (__bf16)f1.x; af[kt][5]  = (__bf16)f1.y;
        af[kt][6]  = (__bf16)f1.z; af[kt][7]  = (__bf16)f1.w;
        af[kt][8]  = (__bf16)f2.x; af[kt][9]  = (__bf16)f2.y;
        af[kt][10] = (__bf16)f2.z; af[kt][11] = (__bf16)f2.w;
        af[kt][12] = (__bf16)f3.x; af[kt][13] = (__bf16)f3.y;
        af[kt][14] = (__bf16)f3.z; af[kt][15] = (__bf16)f3.w;
    }

    // ---- sweep all 16-column tiles
    for (int tn = 0; tn < ntp; ++tn) {
        v8f acc = {};
        const __bf16* wp = Wsw + ((size_t)tn * 32 + lane) * 16;
        #pragma unroll
        for (int kt = 0; kt < NK; ++kt) {
            v16bf bw = *(const v16bf*)(wp + (size_t)kt * ntp * 512);
            acc = __builtin_amdgcn_wmma_f32_16x16x32_bf16(
                false, af[kt], false, bw, (short)0, acc, false, false);
        }
        int cn = (tn << 4) + lr;
        if (cn < Nc) {
            float bv = (bias != nullptr) ? bias[cn] : 0.0f;
            #pragma unroll
            for (int e = 0; e < 8; ++e) {
                int cm = m0 + e + 8 * hf;
                if (cm < M) {
                    float v = acc[e] + bv;
                    if (relu) v = fmaxf(v, 0.0f);
                    C[(size_t)cm * ldc + cn] = v;
                }
            }
        }
    }
}

// Generic fallback (runtime K-step count), one wave per 16x16 tile.
__global__ void gemm_generic_wmma(const float* __restrict__ A, const __bf16* __restrict__ Wsw,
                                  const float* __restrict__ bias, float* __restrict__ C,
                                  int M, int Nc, int ntp, int nk, int ldc, int relu)
{
    int lda = nk * 32;
    int wavesPerBlock = blockDim.x >> 5;
    long long wave = (long long)blockIdx.x * wavesPerBlock + (threadIdx.x >> 5);
    int mt = (M + 15) >> 4;
    if (wave >= (long long)mt * ntp) return;
    int tm = (int)(wave / ntp);
    int tn = (int)(wave % ntp);
    int lane = threadIdx.x & 31;
    int hf = lane >> 4;
    int lr = lane & 15;
    int m0 = tm << 4;
    int am = m0 + lr;
    const float* Arow = A + (size_t)(am < M ? am : 0) * lda;
    const __bf16* wp = Wsw + ((size_t)tn * 32 + lane) * 16;

    v8f acc = {};
    for (int kt = 0; kt < nk; ++kt) {
        const float* p0 = Arow + kt * 32 + 16 * hf;
        float4 f0 = *(const float4*)(p0);
        float4 f1 = *(const float4*)(p0 + 4);
        float4 f2 = *(const float4*)(p0 + 8);
        float4 f3 = *(const float4*)(p0 + 12);
        v16bf af;
        af[0]  = (__bf16)f0.x; af[1]  = (__bf16)f0.y;
        af[2]  = (__bf16)f0.z; af[3]  = (__bf16)f0.w;
        af[4]  = (__bf16)f1.x; af[5]  = (__bf16)f1.y;
        af[6]  = (__bf16)f1.z; af[7]  = (__bf16)f1.w;
        af[8]  = (__bf16)f2.x; af[9]  = (__bf16)f2.y;
        af[10] = (__bf16)f2.z; af[11] = (__bf16)f2.w;
        af[12] = (__bf16)f3.x; af[13] = (__bf16)f3.y;
        af[14] = (__bf16)f3.z; af[15] = (__bf16)f3.w;
        v16bf bw = *(const v16bf*)(wp + (size_t)kt * ntp * 512);
        acc = __builtin_amdgcn_wmma_f32_16x16x32_bf16(
            false, af, false, bw, (short)0, acc, false, false);
    }
    int cn = (tn << 4) + lr;
    if (cn < Nc) {
        float bv = (bias != nullptr) ? bias[cn] : 0.0f;
        #pragma unroll
        for (int e = 0; e < 8; ++e) {
            int cm = m0 + e + 8 * hf;
            if (cm < M) {
                float v = acc[e] + bv;
                if (relu) v = fmaxf(v, 0.0f);
                C[(size_t)cm * ldc + cn] = v;
            }
        }
    }
}

// ---------------------------------------------------------------------------
// Host side
// ---------------------------------------------------------------------------
static inline unsigned zgrid(long long n) {
    long long b = (n + 255) / 256;
    if (b > 16384) b = 16384;
    if (b < 1) b = 1;
    return (unsigned)b;
}
static inline int ceil32(int x) { return (x + 31) & ~31; }
static inline size_t align16(size_t x) { return (x + 15) & ~(size_t)15; }

extern "C" void kernel_launch(void* const* d_in, const int* in_sizes, int n_in,
                              void* d_out, int out_size, void* d_ws, size_t ws_size,
                              hipStream_t stream)
{
    (void)n_in; (void)ws_size;

    const float* pkt = (const float*)d_in[0];
    const float* arv = (const float*)d_in[1];
    const int*   src = (const int*)d_in[2];
    const int*   dst = (const int*)d_in[3];
    const int*   gid = (const int*)d_in[4];
    const float* Wep = (const float*)d_in[6];
    const float* bep = (const float*)d_in[7];
    const float* Wea = (const float*)d_in[8];
    const float* bea = (const float*)d_in[9];
    const float* W0  = (const float*)d_in[10];
    const float* b0  = (const float*)d_in[11];
    const float* W1  = (const float*)d_in[12];
    const float* b1  = (const float*)d_in[13];
    const float* Wc  = (const float*)d_in[14];
    const float* bc  = (const float*)d_in[15];

    const int N   = in_sizes[4];
    const int E   = in_sizes[2];
    const int L   = in_sizes[7];
    const int RAW = in_sizes[6] / L;        // 256 (multiple of 32)
    const int D1  = in_sizes[11];           // 160 (multiple of 32)
    const int D2  = in_sizes[13];           // 200
    const int Cc  = in_sizes[15];           // 55
    const int G   = out_size / Cc;          // 2048
    const int F   = 2 * D2;                 // 400

    const int Lp = ceil32(L);               // 128: padded latent stride
    const int Fp = ceil32(F);               // 416: padded classifier input stride

    // ---- workspace layout (floats), 64B-aligned blocks
    float* ws = (float*)d_ws;
    size_t off = 0;
    auto alloc_f = [&](size_t n) { float* p = ws + off; off = align16(off + n); return p; };
    float* out_is = alloc_f((size_t)N);
    float* in_is  = alloc_f((size_t)N);
    float* cnt    = alloc_f((size_t)G);
    float* feat   = alloc_f((size_t)G * Fp);
    float* t0     = alloc_f((size_t)N * Lp);     // extractor out [N, L] stride Lp
    float* t1     = alloc_f((size_t)N * D1);     // aggregation buffer (max stride D1)
    float* t2     = alloc_f((size_t)N * D1);     // layer-0 out [N, D1]
    float* t3     = alloc_f((size_t)N * D2);     // layer-1 out [N, D2]
    auto alloc_bf = [&](size_t nbf) { __bf16* p = (__bf16*)(ws + off); off = align16(off + (nbf + 1) / 2); return p; };
    const int ntpL = (L + 15) / 16, ntp1 = (D1 + 15) / 16, ntp2 = (D2 + 15) / 16, ntpC = (Cc + 15) / 16;
    __bf16* swEp = alloc_bf((size_t)RAW * ntpL * 16);
    __bf16* swEa = alloc_bf((size_t)RAW * ntpL * 16);
    __bf16* sw0  = alloc_bf((size_t)Lp  * ntp1 * 16);
    __bf16* sw1  = alloc_bf((size_t)D1  * ntp2 * 16);
    __bf16* swC  = alloc_bf((size_t)Fp  * ntpC * 16);

    auto swizzle = [&](const float* W, __bf16* Wsw, int K, int Nc, int Kp) {
        int ntp = (Nc + 15) / 16;
        long long total = (long long)(Kp / 32) * ntp * 512;
        swizzle_w_kernel<<<(unsigned)((total + 255) / 256), 256, 0, stream>>>(W, Wsw, K, Nc, ntp, total);
    };
    auto gemm = [&](const float* A, const __bf16* Wsw, const float* bias, float* Cp,
                    int M, int Kp, int Nc, int ldc, int relu) {
        int ntp = (Nc + 15) / 16;
        int nk = Kp / 32;
        int mt = (M + 15) / 16;
        unsigned blocks = (unsigned)((mt + 7) / 8);    // 8 waves / 256 threads per block
        switch (nk) {
        case 4:  gemm_rows_wmma<4> <<<blocks, 256, 0, stream>>>(A, Wsw, bias, Cp, M, Nc, ntp, ldc, relu); break;
        case 5:  gemm_rows_wmma<5> <<<blocks, 256, 0, stream>>>(A, Wsw, bias, Cp, M, Nc, ntp, ldc, relu); break;
        case 8:  gemm_rows_wmma<8> <<<blocks, 256, 0, stream>>>(A, Wsw, bias, Cp, M, Nc, ntp, ldc, relu); break;
        case 13: gemm_rows_wmma<13><<<blocks, 256, 0, stream>>>(A, Wsw, bias, Cp, M, Nc, ntp, ldc, relu); break;
        default: {
            long long waves = (long long)mt * ntp;
            gemm_generic_wmma<<<(unsigned)((waves + 7) / 8), 256, 0, stream>>>(
                A, Wsw, bias, Cp, M, Nc, ntp, nk, ldc, relu);
        } break;
        }
    };

    // ---- weight swizzles (cheap; deterministic every call)
    swizzle(Wep, swEp, RAW, L, RAW);
    swizzle(Wea, swEa, RAW, L, RAW);
    swizzle(W0,  sw0,  L,   D1, Lp);
    swizzle(W1,  sw1,  D1,  D2, D1);
    swizzle(Wc,  swC,  F,   Cc, Fp);

    // ---- degree normalizers
    zero_kernel<<<zgrid(2LL * N), 256, 0, stream>>>(out_is, 2LL * N);
    degree_kernel<<<(E + 255) / 256, 256, 0, stream>>>(src, dst, out_is, in_is, E);
    rsqrt_kernel<<<(2 * N + 255) / 256, 256, 0, stream>>>(out_is, 2LL * N);

    // ---- pooling prep (feat fully zeroed incl. K-padding columns)
    zero_kernel<<<zgrid(G), 256, 0, stream>>>(cnt, (long long)G);
    zero_kernel<<<zgrid((long long)G * Fp), 256, 0, stream>>>(feat, (long long)G * Fp);
    count_kernel<<<(N + 255) / 256, 256, 0, stream>>>(gid, cnt, N);

    for (int br = 0; br < 2; ++br) {
        const float* x   = (br == 0) ? pkt  : arv;
        const __bf16* we = (br == 0) ? swEp : swEa;
        const float* be  = (br == 0) ? bep  : bea;

        // extractor: relu(x @ We + be) -> t0 [N, L] (stride Lp)
        gemm(x, we, be, t0, N, RAW, L, Lp, 1);

        // GCN layer 0 (in/out-degree norms folded into scatter)
        zero_kernel<<<zgrid((long long)N * Lp), 256, 0, stream>>>(t1, (long long)N * Lp);
        scatter_kernel<<<E, 128, 0, stream>>>(t0, src, dst, out_is, in_is, t1, L, Lp, Lp);
        gemm(t1, sw0, b0, t2, N, Lp, D1, D1, 0);

        // GCN layer 1
        zero_kernel<<<zgrid((long long)N * D1), 256, 0, stream>>>(t1, (long long)N * D1);
        scatter_kernel<<<E, 128, 0, stream>>>(t2, src, dst, out_is, in_is, t1, D1, D1, D1);
        gemm(t1, sw1, b1, t3, N, D1, D2, D2, 0);

        // mean-pool (sum now, divide later) into concatenated [G, F] (stride Fp)
        pool_kernel<<<N, 128, 0, stream>>>(t3, gid, feat, D2, D2, Fp, br * D2);
    }

    divide_kernel<<<(unsigned)(((long long)G * F + 255) / 256), 256, 0, stream>>>(
        feat, cnt, (long long)G * F, F, Fp);

    // classifier: feat [G, F] (stride Fp) @ Wc + bc -> d_out [G, Cc]
    gemm(feat, swC, bc, (float*)d_out, G, Fp, Cc, Cc, 0);
}
